// MultiChev_27462020891069
// MI455X (gfx1250) — compile-verified, matrix operand order checked by reading, not run on hardware
//
#include <hip/hip_runtime.h>

#define NN   100000
#define NE   1600000
#define KD   128
#define OD   400
#define NOUT 1200
#define LDA  136   // padded LDS row stride in halves (multiple of 8 for b128 alignment)

typedef _Float16 half8 __attribute__((ext_vector_type(8)));
typedef _Float16 v16h  __attribute__((ext_vector_type(16)));
typedef float    v8f   __attribute__((ext_vector_type(8)));

// ---------------- utility kernels ----------------

__global__ void zero_f32(float* __restrict__ p, size_t n) {
  size_t i = (size_t)blockIdx.x * blockDim.x + threadIdx.x;
  size_t stride = (size_t)gridDim.x * blockDim.x;
  for (; i < n; i += stride) p[i] = 0.0f;
}

// deg[row] += w  (self-loops removed)
__global__ void deg_kernel(const int* __restrict__ ei, const float* __restrict__ ew,
                           float* __restrict__ deg) {
  int e = blockIdx.x * blockDim.x + threadIdx.x;
  if (e >= NE) return;
  int r = ei[e];
  int c = ei[NE + e];
  if (r != c) atomicAdd(&deg[r], ew[e]);
}

__global__ void dinv_kernel(const float* __restrict__ deg, float* __restrict__ dinv) {
  int i = blockIdx.x * blockDim.x + threadIdx.x;
  if (i >= NN) return;
  float d = deg[i];
  dinv[i] = (d > 0.0f) ? rsqrtf(d) : 0.0f;
}

// lhat_w[e] = -dinv[row] * w * dinv[col]   (w=0 on self loops)
__global__ void lhat_kernel(const int* __restrict__ ei, const float* __restrict__ ew,
                            const float* __restrict__ dinv, float* __restrict__ lw) {
  int e = blockIdx.x * blockDim.x + threadIdx.x;
  if (e >= NE) return;
  int r = ei[e];
  int c = ei[NE + e];
  float w = (r == c) ? 0.0f : ew[e];
  lw[e] = -dinv[r] * w * dinv[c];
}

// out[col] += lhat_w[e] * h[row] : one wave per edge, float4 per lane (128 dims)
__global__ void prop_kernel(const int* __restrict__ ei, const float* __restrict__ lw,
                            const float* __restrict__ h, float* __restrict__ acc) {
  int wid = blockIdx.x * (blockDim.x >> 5) + (threadIdx.x >> 5);
  if (wid >= NE) return;
  float w = lw[wid];
  if (w == 0.0f) return;          // self loops / zero weights contribute nothing
  int lane = threadIdx.x & 31;
  int r = ei[wid];
  int c = ei[NE + wid];
  const float4 v = *(const float4*)(h + (size_t)r * KD + lane * 4);
  float* dst = acc + (size_t)c * KD + lane * 4;
  atomicAdd(dst + 0, w * v.x);
  atomicAdd(dst + 1, w * v.y);
  atomicAdd(dst + 2, w * v.z);
  atomicAdd(dst + 3, w * v.w);
}

// tx2 = 2 * prop(tx1) - tx0
__global__ void tx2_fin_kernel(float* __restrict__ tx2, const float* __restrict__ x) {
  size_t i = (size_t)blockIdx.x * blockDim.x + threadIdx.x;
  if (i >= (size_t)NN * KD) return;
  tx2[i] = 2.0f * tx2[i] - x[i];
}

// Repack 6 weight matrices (row-major [k][n], fp32) into [mat][n][k] split f16 hi/lo.
// Matrix order: 0=W1, 1=W2_0, 2=W2_1, 3=W3_0, 4=W3_1, 5=W3_2
__global__ void wprep_kernel(const float* __restrict__ W1,  const float* __restrict__ W2_0,
                             const float* __restrict__ W2_1, const float* __restrict__ W3_0,
                             const float* __restrict__ W3_1, const float* __restrict__ W3_2,
                             _Float16* __restrict__ Whi, _Float16* __restrict__ Wlo) {
  int idx = blockIdx.x * blockDim.x + threadIdx.x;
  if (idx >= 6 * OD * KD) return;
  int mat = idx / (OD * KD);
  int rem = idx % (OD * KD);
  int n = rem / KD;
  int k = rem % KD;
  const float* W = (mat == 0) ? W1 : (mat == 1) ? W2_0 : (mat == 2) ? W2_1
                 : (mat == 3) ? W3_0 : (mat == 4) ? W3_1 : W3_2;
  float v = W[(size_t)k * OD + n];
  _Float16 h = (_Float16)v;
  Whi[idx] = h;
  Wlo[idx] = (_Float16)(v - (float)h);
}

// ---------------- fused triple-GEMM with split-f16 WMMA ----------------
// Block = 128 threads (4 waves) -> one 16-row slab, 75 column tiles of 16.
// cols [0,400):   tx0@W1
// cols [400,800): tx0@W2_0 + tx1@W2_1
// cols [800,1200):tx0@W3_0 + tx1@W3_1 + tx2@W3_2
__launch_bounds__(128)
__global__ void gemm_kernel(const float* __restrict__ x,   const float* __restrict__ tx1,
                            const float* __restrict__ tx2,
                            const _Float16* __restrict__ Whi, const _Float16* __restrict__ Wlo,
                            const float* __restrict__ b1v, const float* __restrict__ b2v,
                            const float* __restrict__ b3v, float* __restrict__ out) {
  __shared__ __align__(16) _Float16 Ah[3][16][LDA];
  __shared__ __align__(16) _Float16 Al[3][16][LDA];

  const int tid = threadIdx.x;
  const size_t rowbase = (size_t)blockIdx.x * 16;   // 6250 * 16 == 100000 exactly

  // Cooperative load of tx0/tx1/tx2 rows -> LDS, split fp32 into f16 hi + lo.
  #pragma unroll
  for (int s = 0; s < 3; ++s) {
    const float* src = ((s == 0) ? x : (s == 1) ? tx1 : tx2) + rowbase * KD;
    const float4* sp = (const float4*)src;
    for (int i = tid; i < 16 * KD / 4; i += 128) {   // 512 float4
      float4 v = sp[i];
      int row = i >> 5;            // 32 float4 per 128-wide row
      int k = (i & 31) << 2;
      float vv[4] = {v.x, v.y, v.z, v.w};
      #pragma unroll
      for (int j = 0; j < 4; ++j) {
        _Float16 h = (_Float16)vv[j];
        Ah[s][row][k + j] = h;
        Al[s][row][k + j] = (_Float16)(vv[j] - (float)h);
      }
    }
  }
  __syncthreads();

  const int wave = tid >> 5;   // wave32 on gfx1250
  const int lane = tid & 31;
  const int m = lane & 15;     // row (A) / col (B,C,D) index within tile
  const int g = lane >> 4;     // K-half selector for A/B fragments

  for (int t = wave; t < 75; t += 4) {
    const int seg = (t < 25) ? 0 : (t < 50) ? 1 : 2;
    const int n = (t - seg * 25) * 16 + m;                 // column within 400-wide segment
    const int matBase = (seg == 0) ? 0 : (seg == 1) ? 1 : 3;

    v8f acc  = {};   // hi*hi products
    v8f accc = {};   // correction products (independent chain)

    for (int term = 0; term <= seg; ++term) {              // A = tx_term
      const _Float16* wh = Whi + ((size_t)(matBase + term) * OD + n) * KD;
      const _Float16* wl = Wlo + ((size_t)(matBase + term) * OD + n) * KD;
      #pragma unroll
      for (int c = 0; c < 4; ++c) {                        // K chunks of 32
        const int k0 = 32 * c + 8 * g;
        // ISA fragment layout: halves[0..7] = K k0..k0+7, halves[8..15] = K k0+16..k0+23
        half8 ah0 = *(const half8*)&Ah[term][m][k0];
        half8 ah1 = *(const half8*)&Ah[term][m][k0 + 16];
        half8 al0 = *(const half8*)&Al[term][m][k0];
        half8 al1 = *(const half8*)&Al[term][m][k0 + 16];
        half8 bh0 = *(const half8*)(wh + k0);
        half8 bh1 = *(const half8*)(wh + k0 + 16);
        half8 bl0 = *(const half8*)(wl + k0);
        half8 bl1 = *(const half8*)(wl + k0 + 16);
        v16h a_hi = __builtin_shufflevector(ah0, ah1, 0,1,2,3,4,5,6,7,8,9,10,11,12,13,14,15);
        v16h a_lo = __builtin_shufflevector(al0, al1, 0,1,2,3,4,5,6,7,8,9,10,11,12,13,14,15);
        v16h b_hi = __builtin_shufflevector(bh0, bh1, 0,1,2,3,4,5,6,7,8,9,10,11,12,13,14,15);
        v16h b_lo = __builtin_shufflevector(bl0, bl1, 0,1,2,3,4,5,6,7,8,9,10,11,12,13,14,15);
        acc  = __builtin_amdgcn_wmma_f32_16x16x32_f16(false, a_hi, false, b_hi, (short)0, acc,  false, false);
        accc = __builtin_amdgcn_wmma_f32_16x16x32_f16(false, a_hi, false, b_lo, (short)0, accc, false, false);
        accc = __builtin_amdgcn_wmma_f32_16x16x32_f16(false, a_lo, false, b_hi, (short)0, accc, false, false);
      }
    }

    const float* bp = (seg == 0) ? b1v : (seg == 1) ? b2v : b3v;
    const float bias = bp[n];
    // C/D layout: element(v, lane) = D[M = v + 8*g][N = m]
    float* op = out + (rowbase + 8 * g) * NOUT + t * 16 + m;
    #pragma unroll
    for (int v = 0; v < 8; ++v)
      op[(size_t)v * NOUT] = acc[v] + accc[v] + bias;
  }
}

// ---------------- launcher ----------------

extern "C" void kernel_launch(void* const* d_in, const int* in_sizes, int n_in,
                              void* d_out, int out_size, void* d_ws, size_t ws_size,
                              hipStream_t stream) {
  const float* x    = (const float*)d_in[0];
  const int*   ei   = (const int*)d_in[1];   // (2, NE): row = ei[e], col = ei[NE+e]
  const float* ew   = (const float*)d_in[2];
  const float* W1   = (const float*)d_in[3];
  const float* b1   = (const float*)d_in[4];
  const float* W2_0 = (const float*)d_in[5];
  const float* W2_1 = (const float*)d_in[6];
  const float* b2   = (const float*)d_in[7];
  const float* W3_0 = (const float*)d_in[8];
  const float* W3_1 = (const float*)d_in[9];
  const float* W3_2 = (const float*)d_in[10];
  const float* b3   = (const float*)d_in[11];
  float* out = (float*)d_out;

  // Workspace carve-out (~111 MB): deg, dinv, lhat_w, tx1, tx2, packed weights
  char* ws = (char*)d_ws;
  size_t o = 0;
  auto carve = [&](size_t bytes) -> char* {
    char* p = ws + o;
    o = (o + bytes + 255) & ~(size_t)255;
    return p;
  };
  float*    deg  = (float*)carve((size_t)NN * 4);
  float*    dinv = (float*)carve((size_t)NN * 4);
  float*    lw   = (float*)carve((size_t)NE * 4);
  float*    tx1  = (float*)carve((size_t)NN * KD * 4);
  float*    tx2  = (float*)carve((size_t)NN * KD * 4);
  _Float16* Whi  = (_Float16*)carve((size_t)6 * OD * KD * 2);
  _Float16* Wlo  = (_Float16*)carve((size_t)6 * OD * KD * 2);

  // Zero atomic-accumulation targets every call (replay-safe, deterministic work).
  zero_f32<<<1024, 256, 0, stream>>>(deg, (size_t)NN);
  zero_f32<<<4096, 256, 0, stream>>>(tx1, (size_t)NN * KD);
  zero_f32<<<4096, 256, 0, stream>>>(tx2, (size_t)NN * KD);

  wprep_kernel<<<(6 * OD * KD + 255) / 256, 256, 0, stream>>>(
      W1, W2_0, W2_1, W3_0, W3_1, W3_2, Whi, Wlo);

  deg_kernel<<<(NE + 255) / 256, 256, 0, stream>>>(ei, ew, deg);
  dinv_kernel<<<(NN + 255) / 256, 256, 0, stream>>>(deg, dinv);
  lhat_kernel<<<(NE + 255) / 256, 256, 0, stream>>>(ei, ew, dinv, lw);

  // tx1 = L_hat @ x ; tx2_raw = L_hat @ tx1  (8 waves / block, 1 wave per edge)
  prop_kernel<<<(NE + 7) / 8, 256, 0, stream>>>(ei, lw, x, tx1);
  prop_kernel<<<(NE + 7) / 8, 256, 0, stream>>>(ei, lw, tx1, tx2);
  tx2_fin_kernel<<<(int)(((size_t)NN * KD + 255) / 256), 256, 0, stream>>>(tx2, x);

  // Fused [scale_1 | scale_2 | scale_3] GEMM, 6250 row-slabs.
  gemm_kernel<<<NN / 16, 128, 0, stream>>>(x, tx1, tx2, Whi, Wlo, b1, b2, b3, out);
}
